// InnerProductProbe_86912958202203
// MI455X (gfx1250) — matrix operand order repaired and checked by reading, not compiled, not run on hardware
//
#include <hip/hip_runtime.h>
#include <hip/hip_bf16.h>

// Problem constants (B=64, S=512, L=768)
#define BATCH 64
#define SEQ   512
#define LDIM  768
#define MTOT  (BATCH * SEQ)   // 32768 flattened rows of p

typedef __attribute__((ext_vector_type(16))) __bf16 v16bf;
typedef __attribute__((ext_vector_type(8)))  float  v8f;
typedef __attribute__((ext_vector_type(4)))  int    v4i;

union Pack16 { __bf16 h[16]; uint4 u4[2]; };   // 16 bf16 = 32 B
union Frag16 { uint4 u4[2]; v16bf v; };        // WMMA A/B fragment (8 VGPRs)
union Pack8  { uint4 u; __bf16 h[8]; };

#define LDSROW 40   // bf16 stride per LDS tile row: 32 data + 8 pad (80 B, 16B aligned)

// ---------------- CDNA5 async global->LDS (ASYNCcnt) helpers -----------------
#if defined(__has_builtin)
#if __has_builtin(__builtin_amdgcn_global_load_async_to_lds_b128)
#define HAS_ASYNC_LDS 1
#endif
#endif

typedef __attribute__((address_space(1))) v4i* gv4i_p;   // global v4i*
typedef __attribute__((address_space(3))) v4i* lv4i_p;   // LDS v4i*

__device__ __forceinline__ void async_copy_b128(const void* gsrc, void* ldst) {
#ifdef HAS_ASYNC_LDS
  __builtin_amdgcn_global_load_async_to_lds_b128(
      (gv4i_p)(v4i*)(const_cast<void*>(gsrc)),
      (lv4i_p)(v4i*)ldst,
      /*offset=*/0, /*cpol=*/0);
#else
  *(uint4*)ldst = *(const uint4*)gsrc;
#endif
}

__device__ __forceinline__ void async_wait0() {
#ifdef HAS_ASYNC_LDS
#if __has_builtin(__builtin_amdgcn_s_wait_asynccnt)
  __builtin_amdgcn_s_wait_asynccnt(0);
#else
  asm volatile("s_wait_asynccnt 0x0" ::: "memory");
#endif
#endif
}

// ---------------------------------------------------------------------------
// Kernel 1: p[m,g] = sum_h x[m,h] * b[g,h]  (f32 in, bf16 out, f32 accumulate)
// Block tile 128(M) x 64(N), K step 32. 256 threads = 8 waves (4x2 of 32x32).
// Staging is synchronous: must convert f32 -> bf16 in flight.
// ---------------------------------------------------------------------------
__global__ __launch_bounds__(256) void proj_kernel(const float* __restrict__ X,
                                                   const float* __restrict__ Bm,
                                                   __bf16* __restrict__ P) {
  __shared__ __bf16 As[128 * LDSROW];
  __shared__ __bf16 Bs[64 * LDSROW];

  const int nBase   = blockIdx.x * 64;
  const int rowBase = blockIdx.y * 128;
  const int t    = threadIdx.x;
  const int wave = t >> 5;
  const int lane = t & 31;
  const int wm   = wave & 3;            // 4 row groups of 32
  const int wn   = wave >> 2;           // 2 col groups of 32
  const int lrow = lane & 15;
  const int kbA  = (lane < 16) ? 0 : 8;   // A frag K phase
  const int kbB  = (lane < 16) ? 0 : 8;   // B frag K phase

  v8f acc[2][2] = {};

  for (int k0 = 0; k0 < LDIM; k0 += 32) {
    // ---- stage A tile: 128 rows x 32 K (f32 -> bf16) ----
    {
      const int row = t >> 1, half = t & 1;
      const float* src = X + (size_t)(rowBase + row) * LDIM + k0 + half * 16;
      __builtin_prefetch(src + 32, 0, 0);       // global_prefetch_b8 next K tile
      const float4* s4 = (const float4*)src;
      float4 f[4];
#pragma unroll
      for (int q = 0; q < 4; ++q) f[q] = s4[q];
      Pack16 pk;
#pragma unroll
      for (int q = 0; q < 4; ++q) {
        pk.h[q * 4 + 0] = (__bf16)f[q].x; pk.h[q * 4 + 1] = (__bf16)f[q].y;
        pk.h[q * 4 + 2] = (__bf16)f[q].z; pk.h[q * 4 + 3] = (__bf16)f[q].w;
      }
      uint4* dst = (uint4*)&As[row * LDSROW + half * 16];
      dst[0] = pk.u4[0]; dst[1] = pk.u4[1];
    }
    // ---- stage B tile: 64 rows (g) x 32 K (f32 -> bf16) ----
    if (t < 128) {
      const int row = t >> 1, half = t & 1;
      const float* src = Bm + (size_t)(nBase + row) * LDIM + k0 + half * 16;
      const float4* s4 = (const float4*)src;
      float4 f[4];
#pragma unroll
      for (int q = 0; q < 4; ++q) f[q] = s4[q];
      Pack16 pk;
#pragma unroll
      for (int q = 0; q < 4; ++q) {
        pk.h[q * 4 + 0] = (__bf16)f[q].x; pk.h[q * 4 + 1] = (__bf16)f[q].y;
        pk.h[q * 4 + 2] = (__bf16)f[q].z; pk.h[q * 4 + 3] = (__bf16)f[q].w;
      }
      uint4* dst = (uint4*)&Bs[row * LDSROW + half * 16];
      dst[0] = pk.u4[0]; dst[1] = pk.u4[1];
    }
    __syncthreads();

    // ---- fragments per ISA VGPR layout ----
    Frag16 af[2], bfr[2];
#pragma unroll
    for (int i = 0; i < 2; ++i) {
      const __bf16* ar = &As[(wm * 32 + i * 16 + lrow) * LDSROW];
      af[i].u4[0] = *(const uint4*)(ar + kbA);        // K = kbA .. kbA+7
      af[i].u4[1] = *(const uint4*)(ar + 16 + kbA);   // K = 16+kbA .. +7
    }
#pragma unroll
    for (int j = 0; j < 2; ++j) {
      const __bf16* br = &Bs[(wn * 32 + j * 16 + lrow) * LDSROW];
      bfr[j].u4[0] = *(const uint4*)(br + kbB);       // K = kbB .. kbB+7
      bfr[j].u4[1] = *(const uint4*)(br + 16 + kbB);  // K = 16+kbB .. +7
    }
#pragma unroll
    for (int i = 0; i < 2; ++i)
#pragma unroll
      for (int j = 0; j < 2; ++j)
        acc[i][j] = __builtin_amdgcn_wmma_f32_16x16x32_bf16(
            false, af[i].v, false, bfr[j].v, (short)0, acc[i][j], false, false);
    __syncthreads();
  }

  // ---- epilogue: p -> bf16 global ----
  const int rOff = (lane < 16) ? 0 : 8;
#pragma unroll
  for (int i = 0; i < 2; ++i)
#pragma unroll
    for (int j = 0; j < 2; ++j) {
      const int gcol = nBase + wn * 32 + j * 16 + lrow;
#pragma unroll
      for (int r = 0; r < 8; ++r) {
        const int grow = rowBase + wm * 32 + i * 16 + r + rOff;
        P[(size_t)grow * LDIM + gcol] = (__bf16)acc[i][j][r];
      }
    }
}

// ---------------------------------------------------------------------------
// Kernel 2: sq[m] = sum_g p[m,g]^2  (one wave per row, b128 loads)
// ---------------------------------------------------------------------------
__global__ __launch_bounds__(256) void sq_kernel(const __bf16* __restrict__ P,
                                                 float* __restrict__ SQ) {
  const int wave = threadIdx.x >> 5, lane = threadIdx.x & 31;
  const int row = blockIdx.x * 8 + wave;
  const uint4* pr4 = (const uint4*)(P + (size_t)row * LDIM);   // 96 chunks of 8 bf16
  float s = 0.0f;
#pragma unroll
  for (int q = 0; q < 3; ++q) {
    Pack8 pk;
    pk.u = pr4[lane + q * 32];
#pragma unroll
    for (int e = 0; e < 8; ++e) {
      float v = (float)pk.h[e];
      s += v * v;
    }
  }
#pragma unroll
  for (int off = 16; off; off >>= 1) s += __shfl_xor(s, off, 32);
  if (lane == 0) SQ[row] = s;
}

// ---------------------------------------------------------------------------
// Kernel 3: per-batch Gram + fused epilogue, double-buffered async->LDS.
// dist[b,s,t] = sq[s] + sq[t] - 2 * sum_g p[s,g] p[t,g]
// Block tile 128x128, K step 32. 8 waves = 4x2 grid of 32x64 wave tiles.
// ---------------------------------------------------------------------------
__global__ __launch_bounds__(256) void gram_kernel(const __bf16* __restrict__ P,
                                                   const float* __restrict__ SQ,
                                                   float* __restrict__ Out) {
  __shared__ __bf16 As[2][128 * LDSROW];
  __shared__ __bf16 Bs[2][128 * LDSROW];

  const int b  = blockIdx.z;
  const int tM = blockIdx.y, tN = blockIdx.x;
  const int rowBase = b * SEQ + tM * 128;   // global p row for M (s)
  const int colBase = b * SEQ + tN * 128;   // global p row for N (t)
  const int t    = threadIdx.x;
  const int wave = t >> 5;
  const int lane = t & 31;
  const int wm   = wave & 3;      // 4 groups of 32 rows
  const int wn   = wave >> 2;     // 2 groups of 64 cols
  const int lrow = lane & 15;
  const int kbA  = (lane < 16) ? 0 : 8;
  const int kbB  = (lane < 16) ? 0 : 8;

  const int srow = t >> 1, shalf = t & 1;   // staging role: 32 B per thread per tile

  // Stage one 128x32 A tile + 128x32 B tile (bf16) into LDS buffer `buf`
  // via async global->LDS b128 transfers (4 per thread).
  auto stage = [&](int buf, int k0) {
    const char* sa = (const char*)(P + (size_t)(rowBase + srow) * LDIM + k0 + shalf * 16);
    const char* sb = (const char*)(P + (size_t)(colBase + srow) * LDIM + k0 + shalf * 16);
    char* da = (char*)&As[buf][srow * LDSROW + shalf * 16];
    char* db = (char*)&Bs[buf][srow * LDSROW + shalf * 16];
    async_copy_b128(sa,      da);
    async_copy_b128(sa + 16, da + 16);
    async_copy_b128(sb,      db);
    async_copy_b128(sb + 16, db + 16);
  };

  v8f acc[2][4] = {};

  stage(0, 0);
  async_wait0();
  __syncthreads();

  const int NT = LDIM / 32;   // 24 K tiles
  for (int kt = 0; kt < NT; ++kt) {
    const int cur = kt & 1;

    // ---- fragments from current buffer ----
    Frag16 af[2], bfr[4];
#pragma unroll
    for (int i = 0; i < 2; ++i) {
      const __bf16* ar = &As[cur][(wm * 32 + i * 16 + lrow) * LDSROW];
      af[i].u4[0] = *(const uint4*)(ar + kbA);
      af[i].u4[1] = *(const uint4*)(ar + 16 + kbA);
    }
#pragma unroll
    for (int j = 0; j < 4; ++j) {
      const __bf16* br = &Bs[cur][(wn * 64 + j * 16 + lrow) * LDSROW];
      bfr[j].u4[0] = *(const uint4*)(br + kbB);
      bfr[j].u4[1] = *(const uint4*)(br + 16 + kbB);
    }

    // ---- prefetch next tile into the alternate buffer while WMMAs run ----
    if (kt + 1 < NT) stage(1 - cur, (kt + 1) * 32);

#pragma unroll
    for (int i = 0; i < 2; ++i)
#pragma unroll
      for (int j = 0; j < 4; ++j)
        acc[i][j] = __builtin_amdgcn_wmma_f32_16x16x32_bf16(
            false, af[i].v, false, bfr[j].v, (short)0, acc[i][j], false, false);

    async_wait0();      // own async transfers complete
    __syncthreads();    // all waves' transfers visible; cur buffer free to overwrite
  }

  // ---- fused epilogue ----
  const int rOff = (lane < 16) ? 0 : 8;
#pragma unroll
  for (int j = 0; j < 4; ++j) {
    const int tcol = tN * 128 + wn * 64 + j * 16 + lrow;       // t in [0,512)
    const float sqt = SQ[b * SEQ + tcol];
#pragma unroll
    for (int i = 0; i < 2; ++i)
#pragma unroll
      for (int r = 0; r < 8; ++r) {
        const int sr = tM * 128 + wm * 32 + i * 16 + r + rOff; // s in [0,512)
        const float sqs = SQ[b * SEQ + sr];
        Out[((size_t)b * SEQ + sr) * SEQ + tcol] = sqs + sqt - 2.0f * acc[i][j][r];
      }
  }
}

// ---------------------------------------------------------------------------
extern "C" void kernel_launch(void* const* d_in, const int* in_sizes, int n_in,
                              void* d_out, int out_size, void* d_ws, size_t ws_size,
                              hipStream_t stream) {
  const float* x  = (const float*)d_in[0];   // [64,512,768] f32
  const float* bm = (const float*)d_in[1];   // [768,768]    f32
  float* out = (float*)d_out;                // [64,512,512] f32

  // Workspace layout: p bf16 [32768,768] then sq f32 [32768]
  __bf16* pb = (__bf16*)d_ws;
  float*  sq = (float*)((char*)d_ws + (size_t)MTOT * LDIM * sizeof(__bf16));

  dim3 g1(LDIM / 64, MTOT / 128, 1);         // 12 x 256
  proj_kernel<<<g1, 256, 0, stream>>>(x, bm, pb);

  sq_kernel<<<MTOT / 8, 256, 0, stream>>>(pb, sq);

  dim3 g2(SEQ / 128, SEQ / 128, BATCH);      // 4 x 4 x 64
  gram_kernel<<<g2, 256, 0, stream>>>(pb, sq, out);
}
